// Particle_filter_89464168776343
// MI455X (gfx1250) — matrix operand order repaired and checked by reading, not compiled, not run on hardware
//
#include <hip/hip_runtime.h>
#include <hip/hip_bf16.h>
#include <math.h>

typedef __attribute__((ext_vector_type(16))) _Float16 v16h;
typedef __attribute__((ext_vector_type(8)))  float    v8f;

#define TM 32
#define TN 128
#define TK 32
#define BROW 40          // padded LDS row stride in halves (80 B -> conflict-free)
#define NSTAGE 3
#define BN_ROWS 512

// ---------------------------------------------------------------------------
// CDNA5 async global->LDS copy (guarded; falls back to sync register copy)
// ---------------------------------------------------------------------------
#if defined(__HIP_DEVICE_COMPILE__) && \
    __has_builtin(__builtin_amdgcn_global_load_async_to_lds_b128) && \
    __has_builtin(__builtin_amdgcn_global_load_async_to_lds_b64)
#define USE_ASYNC 1
#else
#define USE_ASYNC 0
#endif

#define AS1 __attribute__((address_space(1)))
#define AS3 __attribute__((address_space(3)))

typedef int g_v4i __attribute__((vector_size(16)));
typedef int g_v2i __attribute__((vector_size(8)));

__device__ __forceinline__ void copy_b128(_Float16* l, const _Float16* g) {
#if USE_ASYNC
    __builtin_amdgcn_global_load_async_to_lds_b128((AS1 g_v4i*)g, (AS3 g_v4i*)l, 0, 0);
#else
    ((uint4*)l)[0] = ((const uint4*)g)[0];
#endif
}
__device__ __forceinline__ void copy_b64(_Float16* l, const _Float16* g) {
#if USE_ASYNC
    __builtin_amdgcn_global_load_async_to_lds_b64((AS1 g_v2i*)g, (AS3 g_v2i*)l, 0, 0);
#else
    ((uint2*)l)[0] = ((const uint2*)g)[0];
#endif
}
// wait until at most N async ops remain outstanding (per-wave, in-order)
#if USE_ASYNC
#if __has_builtin(__builtin_amdgcn_s_wait_asynccnt)
#define WAIT_ASYNC(N) __builtin_amdgcn_s_wait_asynccnt(N)
#else
#define WAIT_ASYNC(N) asm volatile("s_wait_asynccnt " #N ::: "memory")
#endif
#else
#define WAIT_ASYNC(N) ((void)0)
#endif

// ---------------------------------------------------------------------------
// f32 -> f16 cast (image / activations side)
// ---------------------------------------------------------------------------
__global__ __launch_bounds__(256) void cast_f32_f16(const float* __restrict__ in,
                                                    _Float16* __restrict__ out, int n) {
    int i = blockIdx.x * 256 + threadIdx.x;
    if (i < n) out[i] = (_Float16)in[i];
}

// f32 KxN weights -> f16 transposed N x Kpad (K zero-padded to mult of 32)
__global__ __launch_bounds__(256)
void cast_transpose_f16(const float* __restrict__ w, _Float16* __restrict__ wT,
                        int K, int N, int Kpad) {
    int i = blockIdx.x * 256 + threadIdx.x;
    if (i >= N * Kpad) return;
    int n = i / Kpad;
    int k = i - n * Kpad;
    wT[i] = (k < K) ? (_Float16)w[(size_t)k * N + n] : (_Float16)0.0f;
}

// ---------------------------------------------------------------------------
// Implicit-GEMM conv / FC, WMMA f32 <- f16 x f16.
// 3-stage async-LDS pipeline: issue stage ks+2, wait asynccnt<=3 so stage ks+1
// is resident while the newest copy stays in flight a full iteration.
// Weights pre-transposed wT[n*Kpad+k]; all LDS fragment loads are b128.
// Workgroup 256 thr / 8 waves, tile 32(M) x 128(N), 2 WMMA / wave / K-step.
// ---------------------------------------------------------------------------
__global__ __launch_bounds__(256)
void conv_wmma(const _Float16* __restrict__ in, const _Float16* __restrict__ wT,
               const float* __restrict__ bias, _Float16* __restrict__ out,
               int IH, int IW, int IC, int OH, int OW, int OC,
               int KH, int KW, int SH, int SW, int Kpad)
{
    __shared__ __align__(16) _Float16 As[NSTAGE][TM][BROW];   // 3 x 2.5 KB
    __shared__ __align__(16) _Float16 Bs[NSTAGE][TN][BROW];   // 3 x 10  KB

    const int K   = KH * KW * IC;
    const int N   = OC;
    const int tid = threadIdx.x;
    const int m0  = blockIdx.y * TM;
    const int n0  = blockIdx.x * TN;

    // ---- A staging coords: thread -> (row, 4 consecutive k) ----
    const int a_row = tid >> 3;
    const int a_kk  = (tid & 7) * 4;
    const int m     = m0 + a_row;
    const int ohw   = OH * OW;
    const int bb    = m / ohw;
    const int rem   = m - bb * ohw;
    const int oh    = rem / OW;
    const int ow    = rem - oh * OW;
    const int IWIC  = IW * IC;
    const size_t base0 = (((size_t)bb * IH + oh * SH) * IW + ow * SW) * IC;
    const int kwic  = KW * IC;
    const bool fast = ((IC & 3) == 0);
    const bool k1x1 = (KH == 1 && KW == 1);

    // ---- B staging coords: thread -> (n-row, 16-half segment) ----
    const int bn_row = tid >> 1;           // 0..127
    const int b_seg  = (tid & 1) * 16;     // 0 / 16

    // ---- wave / lane layout ----
    const int wave = tid >> 5;
    const int lane = tid & 31;
    const int wr   = wave >> 2;            // M 0..1
    const int wc   = wave & 3;             // N 0..3
    const int lhi  = (lane < 16) ? 0 : 1;
    const int l15  = lane & 15;
    const int kb0  = lhi ? 16 : 0;         // A frag K base (contiguous, matches B)
    const int koff = lhi ? 16 : 0;         // B frag K base

    v8f acc0 = {};
    v8f acc1 = {};

    const int ksteps = Kpad / TK;

    auto stage = [&](int ks, int s) {
        const int k0 = ks * TK;
        // B tile: 2 async b128 per thread, N-major / K-contiguous, no tail
        {
            const _Float16* g = wT + (size_t)(n0 + bn_row) * Kpad + k0 + b_seg;
            _Float16* l = &Bs[s][bn_row][b_seg];
            copy_b128(l, g);
            copy_b128(l + 8, g + 8);
        }
        // A tile (im2col)
        {
            const int kb = k0 + a_kk;
            _Float16* l = &As[s][a_row][a_kk];
            if (fast) {                    // 4 halves share one (kh,kw): 1 async b64
                int kh = 0, kw = 0, icb = kb;
                if (!k1x1) {
                    kh = kb / kwic;
                    int r2 = kb - kh * kwic;
                    kw = r2 / IC;
                    icb = r2 - kw * IC;
                }
                copy_b64(l, in + base0 + (size_t)kh * IWIC + kw * IC + icb);
            } else {                       // conv1 (IC=3): scalar gather + K tail
                union { _Float16 h[4]; uint2 u; } t;
                #pragma unroll
                for (int i = 0; i < 4; ++i) {
                    const int k = kb + i;
                    _Float16 v = (_Float16)0.0f;
                    if (k < K) {
                        const int kh = k / kwic;
                        const int r2 = k - kh * kwic;
                        const int kw = r2 / IC;
                        const int ic = r2 - kw * IC;
                        v = in[base0 + (size_t)kh * IWIC + kw * IC + ic];
                    }
                    t.h[i] = v;
                }
                ((uint2*)l)[0] = t.u;
            }
        }
    };

    // wait for all but the newest stage (3 async ops/stage fast, 2 slow)
    auto wait_keep1 = [&]() {
        if (fast) WAIT_ASYNC(3);
        else      WAIT_ASYNC(2);
    };

    // ---- prologue: prime two stages ----
    stage(0, 0);
    if (ksteps > 1) {
        stage(1, 1);
        wait_keep1();          // stage 0 resident, stage 1 in flight
    } else {
        WAIT_ASYNC(0);
    }
    __syncthreads();

    int s = 0;
    for (int ks = 0; ks < ksteps; ++ks) {
        if (ks + 2 < ksteps) {
            int s2 = s + 2; if (s2 >= NSTAGE) s2 -= NSTAGE;
            stage(ks + 2, s2);
        }

        // ---- fragments: all 16-byte LDS vector loads ----
        v16h afrag;
        {
            const _Float16* ap = &As[s][wr * 16 + l15][kb0];
            union { uint4 q[2]; v16h v; } u;
            u.q[0] = ((const uint4*)ap)[0];        // K kb0..kb0+7
            u.q[1] = ((const uint4*)(ap + 8))[0];  // K kb0+8..kb0+15
            afrag = u.v;
        }
        v16h bfrag0, bfrag1;
        {
            const _Float16* bp0 = &Bs[s][wc * 32 + l15][koff];
            const _Float16* bp1 = &Bs[s][wc * 32 + 16 + l15][koff];
            union { uint4 q[2]; v16h v; } u0, u1;
            u0.q[0] = ((const uint4*)bp0)[0];
            u0.q[1] = ((const uint4*)bp0)[1];
            u1.q[0] = ((const uint4*)bp1)[0];
            u1.q[1] = ((const uint4*)bp1)[1];
            bfrag0 = u0.v;
            bfrag1 = u1.v;
        }

        acc0 = __builtin_amdgcn_wmma_f32_16x16x32_f16(false, afrag, false, bfrag0,
                                                      (short)0, acc0, false, false);
        acc1 = __builtin_amdgcn_wmma_f32_16x16x32_f16(false, afrag, false, bfrag1,
                                                      (short)0, acc1, false, false);

        // ensure the stage the NEXT iteration reads is resident
        if (ks + 2 < ksteps) wait_keep1();
        else                 WAIT_ASYNC(0);
        __syncthreads();

        ++s; if (s == NSTAGE) s = 0;
    }

    // ---- epilogue: bias + relu, f16 row-major MxN (NHWC) ----
    const int colA = n0 + wc * 32 + l15;
    const int colB = colA + 16;
    const float bA = bias[colA];
    const float bB = bias[colB];
    #pragma unroll
    for (int vv = 0; vv < 8; ++vv) {
        const int mrow = m0 + wr * 16 + vv + (lhi ? 8 : 0);
        float x0 = fmaxf(acc0[vv] + bA, 0.0f);
        float x1 = fmaxf(acc1[vv] + bB, 0.0f);
        out[(size_t)mrow * N + colA] = (_Float16)x0;
        out[(size_t)mrow * N + colB] = (_Float16)x1;
    }
}

// ---------------------------------------------------------------------------
// BatchNorm: deterministic, coalesced. blockDim == OC, thread owns channel.
// ---------------------------------------------------------------------------
__global__ __launch_bounds__(512)
void bn_partial(const _Float16* __restrict__ act, int M, int OC,
                float* __restrict__ psum, float* __restrict__ psumsq)
{
    const int c  = threadIdx.x;
    const int r0 = blockIdx.x * BN_ROWS;
    const int r1 = (r0 + BN_ROWS < M) ? r0 + BN_ROWS : M;
    float s = 0.0f, q = 0.0f;
    for (int r = r0; r < r1; ++r) {
        float v = (float)act[(size_t)r * OC + c];
        s += v; q += v * v;
    }
    psum[blockIdx.x * OC + c]   = s;
    psumsq[blockIdx.x * OC + c] = q;
}

__global__ __launch_bounds__(512)
void bn_finalize(const float* __restrict__ psum, const float* __restrict__ psumsq,
                 int nchunk, int OC, int M,
                 float* __restrict__ mean, float* __restrict__ rstd)
{
    const int c = threadIdx.x;
    float s = 0.0f, q = 0.0f;
    for (int ch = 0; ch < nchunk; ++ch) { s += psum[ch * OC + c]; q += psumsq[ch * OC + c]; }
    const float mn = s / (float)M;
    mean[c] = mn;
    rstd[c] = rsqrtf(q / (float)M - mn * mn + 1e-3f);
}

__global__ __launch_bounds__(256)
void bn_apply(_Float16* __restrict__ act, int total, int ocmask,
              const float* __restrict__ mean, const float* __restrict__ rstd)
{
    int i = blockIdx.x * 256 + threadIdx.x;
    if (i < total) {
        int c = i & ocmask;
        float v = (float)act[i];
        act[i] = (_Float16)((v - mean[c]) * rstd[c]);
    }
}

// ---------------------------------------------------------------------------
// Likelihood head (repeat over P collapses to per-batch scalar)
// ---------------------------------------------------------------------------
__global__ __launch_bounds__(128)
void lk_head(const _Float16* __restrict__ enc,
             const float* __restrict__ lk1_w, const float* __restrict__ lk1_b,
             const float* __restrict__ lk2_w, const float* __restrict__ lk2_b,
             const float* __restrict__ lk3_w, const float* __restrict__ lk3_b,
             float* __restrict__ lval)
{
    __shared__ float e[128];
    __shared__ float h1[128];
    __shared__ float h2[32];
    const int b = blockIdx.x;
    const int t = threadIdx.x;
    e[t] = (float)enc[b * 128 + t];
    __syncthreads();
    {
        float a = lk1_b[t];
        for (int i = 0; i < 128; ++i) a += e[i] * lk1_w[i * 128 + t];
        h1[t] = fmaxf(a, 0.0f);
    }
    __syncthreads();
    if (t < 32) {
        float a = lk2_b[t];
        for (int i = 0; i < 128; ++i) a += h1[i] * lk2_w[i * 32 + t];
        h2[t] = fmaxf(a, 0.0f);
    }
    __syncthreads();
    if (t == 0) {
        float a = lk3_b[0];
        for (int i = 0; i < 32; ++i) a += h2[i] * lk3_w[i];
        lval[b] = 1.0f / (1.0f + expf(-a));
    }
}

__global__ __launch_bounds__(256)
void like_fill(const float* __restrict__ lval, float* __restrict__ out)
{
    int i = blockIdx.x * 256 + threadIdx.x;      // 32768 = B*P
    if (i < 32 * 1024) {
        float c = lval[i >> 10];
        out[i] = c / (c * 1024.0f);              // exact normalized likelihood
    }
}

// ---------------------------------------------------------------------------
// Particle propagation: trig + tiny MLP (2->32->64->2), weights in LDS
// ---------------------------------------------------------------------------
__global__ __launch_bounds__(256)
void particle_step(const float* __restrict__ parts,
                   const float* __restrict__ pr1_w, const float* __restrict__ pr1_b,
                   const float* __restrict__ pr2_w, const float* __restrict__ pr2_b,
                   const float* __restrict__ pr3_w, const float* __restrict__ pr3_b,
                   float* __restrict__ out)
{
    __shared__ float w1s[64], b1s[32], w2s[2048], b2s[64], w3s[128], b3s[2];
    for (int i = threadIdx.x; i < 64;   i += 256) w1s[i] = pr1_w[i];
    for (int i = threadIdx.x; i < 32;   i += 256) b1s[i] = pr1_b[i];
    for (int i = threadIdx.x; i < 2048; i += 256) w2s[i] = pr2_w[i];
    for (int i = threadIdx.x; i < 64;   i += 256) b2s[i] = pr2_b[i];
    for (int i = threadIdx.x; i < 128;  i += 256) w3s[i] = pr3_w[i];
    if (threadIdx.x < 2) b3s[threadIdx.x] = pr3_b[threadIdx.x];
    __syncthreads();

    const int p = blockIdx.x * 256 + threadIdx.x;
    if (p >= 32 * 1024) return;
    const float* s = parts + (size_t)p * 5;
    const float s0 = s[0], s1 = s[1], s2 = s[2], s3 = s[3], s4 = s[4];
    const float theta = s2 + s4;
    const float v = s3, td = s4;
    const float x = s0 + v * sinf(theta);
    const float y = s1 + v * cosf(theta);

    float h1[32];
    #pragma unroll
    for (int j = 0; j < 32; ++j)
        h1[j] = fmaxf(w1s[j] * v + w1s[32 + j] * td + b1s[j], 0.0f);
    float h2[64];
    for (int j = 0; j < 64; ++j) {
        float a = b2s[j];
        #pragma unroll
        for (int i = 0; i < 32; ++i) a += h1[i] * w2s[i * 64 + j];
        h2[j] = fmaxf(a, 0.0f);
    }
    float u0 = b3s[0], u1 = b3s[1];
    for (int i = 0; i < 64; ++i) { u0 += h2[i] * w3s[i * 2]; u1 += h2[i] * w3s[i * 2 + 1]; }
    u0 = tanhf(u0); u1 = tanhf(u1);

    float* o = out + (size_t)p * 5;
    o[0] = x; o[1] = y; o[2] = theta; o[3] = v + u0; o[4] = td + u1;
}

// ---------------------------------------------------------------------------
// Host side
// ---------------------------------------------------------------------------
static inline void launch_castT(const float* w, _Float16* wT, int K, int N, int Kpad,
                                hipStream_t st) {
    int n = N * Kpad;
    cast_transpose_f16<<<(n + 255) / 256, 256, 0, st>>>(w, wT, K, N, Kpad);
}

static inline void launch_conv(const _Float16* in, const _Float16* wT, const float* bias,
                               _Float16* out, int Bn, int IH, int IW, int IC,
                               int OH, int OW, int OC, int KH, int KW, int SH, int SW,
                               int Kpad, hipStream_t st) {
    const int M = Bn * OH * OW;
    dim3 grid(OC / TN, M / TM);
    conv_wmma<<<grid, 256, 0, st>>>(in, wT, bias, out, IH, IW, IC, OH, OW, OC,
                                    KH, KW, SH, SW, Kpad);
}

static inline void launch_bn(_Float16* act, int M, int OC, float* mean, float* rstd,
                             float* psum, float* psumsq, hipStream_t st) {
    const int nchunk = (M + BN_ROWS - 1) / BN_ROWS;
    bn_partial<<<nchunk, OC, 0, st>>>(act, M, OC, psum, psumsq);
    bn_finalize<<<1, OC, 0, st>>>(psum, psumsq, nchunk, OC, M, mean, rstd);
    const int total = M * OC;
    bn_apply<<<(total + 255) / 256, 256, 0, st>>>(act, total, OC - 1, mean, rstd);
}

extern "C" void kernel_launch(void* const* d_in, const int* in_sizes, int n_in,
                              void* d_out, int out_size, void* d_ws, size_t ws_size,
                              hipStream_t stream)
{
    const float* image  = (const float*)d_in[0];
    const float* parts  = (const float*)d_in[1];
    const float* w1  = (const float*)d_in[2];  const float* b1  = (const float*)d_in[3];
    const float* w2  = (const float*)d_in[4];  const float* b2  = (const float*)d_in[5];
    const float* w3  = (const float*)d_in[6];  const float* b3  = (const float*)d_in[7];
    const float* w4  = (const float*)d_in[8];  const float* b4  = (const float*)d_in[9];
    const float* w5  = (const float*)d_in[10]; const float* b5  = (const float*)d_in[11];
    const float* fc1w = (const float*)d_in[12]; const float* fc1b = (const float*)d_in[13];
    const float* fc2w = (const float*)d_in[14]; const float* fc2b = (const float*)d_in[15];
    const float* fc4w = (const float*)d_in[16]; const float* fc4b = (const float*)d_in[17];
    const float* lk1w = (const float*)d_in[18]; const float* lk1b = (const float*)d_in[19];
    const float* lk2w = (const float*)d_in[20]; const float* lk2b = (const float*)d_in[21];
    const float* lk3w = (const float*)d_in[22]; const float* lk3b = (const float*)d_in[23];
    const float* pr1w = (const float*)d_in[24]; const float* pr1b = (const float*)d_in[25];
    const float* pr2w = (const float*)d_in[26]; const float* pr2b = (const float*)d_in[27];
    const float* pr3w = (const float*)d_in[28]; const float* pr3b = (const float*)d_in[29];

    float* out_state = (float*)d_out;              // 32*1024*5 = 163840
    float* out_like  = (float*)d_out + 163840;     // 32*1024   = 32768

    // ---- workspace carve-up ----
    _Float16* ws16 = (_Float16*)d_ws;
    size_t off = 0;
    auto take = [&](size_t n) -> _Float16* {
        _Float16* p = ws16 + off;
        off += (n + 255) & ~(size_t)255;
        return p;
    };
    _Float16* img_h = take(32u * 224u * 224u * 3u);   // 4,816,896
    _Float16* w1T   = take(128 * 160);                // Kpad 160
    _Float16* w2T   = take(128 * 3200);
    _Float16* w3T   = take(256 * 1152);
    _Float16* w4T   = take(256 * 2304);
    _Float16* w5T   = take(512 * 2304);
    _Float16* fc1T  = take(1024u * 24576u);
    _Float16* fc2T  = take(512 * 1024);
    _Float16* fc4T  = take(128 * 512);
    _Float16* bufA  = take(32u * 73u * 73u * 128u);   // 21,827,584 (max act)
    _Float16* bufB  = take(32u * 69u * 35u * 128u);   // 9,891,840
    float* fstat   = (float*)(ws16 + off);
    float* bn_mean = fstat;
    float* bn_rstd = fstat + 512;
    float* lval    = fstat + 1024;
    float* psum    = fstat + 2048;                    // 400*512 max
    float* psumsq  = psum + 400 * 512;

    // ---- casts (weights transposed + K-padded) ----
    {
        int n = 32 * 224 * 224 * 3;
        cast_f32_f16<<<(n + 255) / 256, 256, 0, stream>>>(image, img_h, n);
    }
    launch_castT(w1,  w1T,  147,   128,  160,   stream);
    launch_castT(w2,  w2T,  3200,  128,  3200,  stream);
    launch_castT(w3,  w3T,  1152,  256,  1152,  stream);
    launch_castT(w4,  w4T,  2304,  256,  2304,  stream);
    launch_castT(w5,  w5T,  2304,  512,  2304,  stream);
    launch_castT(fc1w, fc1T, 24576, 1024, 24576, stream);
    launch_castT(fc2w, fc2T, 1024,  512,  1024,  stream);
    launch_castT(fc4w, fc4T, 512,   128,  512,   stream);

    // ---- conv stack: WMMA conv+bias+relu, then deterministic BN in place ----
    launch_conv(img_h, w1T, b1, bufA, 32, 224, 224, 3, 73, 73, 128, 7, 7, 3, 3, 160, stream);
    launch_bn(bufA, 32 * 73 * 73, 128, bn_mean, bn_rstd, psum, psumsq, stream);
    launch_conv(bufA, w2T, b2, bufB, 32, 73, 73, 128, 69, 35, 128, 5, 5, 1, 2, 3200, stream);
    launch_bn(bufB, 32 * 69 * 35, 128, bn_mean, bn_rstd, psum, psumsq, stream);
    launch_conv(bufB, w3T, b3, bufA, 32, 69, 35, 128, 67, 17, 256, 3, 3, 1, 2, 1152, stream);
    launch_bn(bufA, 32 * 67 * 17, 256, bn_mean, bn_rstd, psum, psumsq, stream);
    launch_conv(bufA, w4T, b4, bufB, 32, 67, 17, 256, 33, 8, 256, 3, 3, 2, 2, 2304, stream);
    launch_bn(bufB, 32 * 33 * 8, 256, bn_mean, bn_rstd, psum, psumsq, stream);
    launch_conv(bufB, w5T, b5, bufA, 32, 33, 8, 256, 16, 3, 512, 3, 3, 2, 2, 2304, stream);
    launch_bn(bufA, 32 * 16 * 3, 512, bn_mean, bn_rstd, psum, psumsq, stream);

    // ---- FC encoder (repeat over E collapses) ----
    _Float16* h1  = bufB;               // 32 x 1024
    _Float16* h2  = bufB + 32768;       // 32 x 512
    _Float16* enc = bufB + 49152;       // 32 x 128
    launch_conv(bufA, fc1T, fc1b, h1,  32, 1, 1, 24576, 1, 1, 1024, 1, 1, 1, 1, 24576, stream);
    launch_conv(h1,   fc2T, fc2b, h2,  32, 1, 1, 1024,  1, 1, 512,  1, 1, 1, 1, 1024,  stream);
    launch_conv(h2,   fc4T, fc4b, enc, 32, 1, 1, 512,   1, 1, 128,  1, 1, 1, 1, 512,   stream);

    // ---- likelihood head + broadcast ----
    lk_head<<<32, 128, 0, stream>>>(enc, lk1w, lk1b, lk2w, lk2b, lk3w, lk3b, lval);
    like_fill<<<128, 256, 0, stream>>>(lval, out_like);

    // ---- particle propagation ----
    particle_step<<<128, 256, 0, stream>>>(parts, pr1w, pr1b, pr2w, pr2b, pr3w, pr3b,
                                           out_state);
}
